// Dinov3VitAttention_90718299226613
// MI455X (gfx1250) — compile-verified, matrix operand order checked by reading, not run on hardware
//
#include <hip/hip_runtime.h>
#include <hip/hip_bf16.h>

typedef __bf16 bf16;
typedef __attribute__((ext_vector_type(16))) bf16 v16bf;
typedef __attribute__((ext_vector_type(8)))  bf16 v8bf;
typedef __attribute__((ext_vector_type(8)))  float v8f;

#define WMMA_BF16(a, b, c) \
  __builtin_amdgcn_wmma_f32_16x16x32_bf16(false, (a), false, (b), (short)0, (c), false, false)

// ---------------------------------------------------------------------------
// Fragment loader for 16-bit A/B matrices of v_wmma_*_16x16x32_bf16.
// Lane (r, hi): elements 0..7 hold K = k0 + hi*8 .. +7,
// elements 8..15 hold K = k0 + 16 + hi*8 .. +7 of row `row`.
// Both chunks are 16B-contiguous -> two b128 loads (global or LDS).
// ---------------------------------------------------------------------------
__device__ __forceinline__ v16bf load_frag(const bf16* base, int stride, int row,
                                           int k0, int hi) {
  const bf16* p = base + (size_t)row * stride + k0 + hi * 8;
  v8bf lo = *(const v8bf*)(p);
  v8bf hi8 = *(const v8bf*)(p + 16);
  v16bf out;
#pragma unroll
  for (int i = 0; i < 8; ++i) {
    out[i] = lo[i];
    out[i + 8] = hi8[i];
  }
  return out;
}

// ---------------------------------------------------------------------------
// Elementwise fp32 -> bf16
// ---------------------------------------------------------------------------
__global__ __launch_bounds__(256) void cvt_f32_bf16(const float* __restrict__ in,
                                                    bf16* __restrict__ out, int n) {
  int i = blockIdx.x * 256 + threadIdx.x;
  if (i < n) out[i] = (bf16)in[i];
}

// ---------------------------------------------------------------------------
// GEMM: C[m,n] = sum_k A[m,k] * W[n,k] + bias[n]   (einsum 'bnd,ed->bne')
// A: [M,K] bf16 row-major, W: [N,K] bf16 row-major, C: [M,N] fp32.
// Workgroup: 256 thr = 8 waves arranged 4(M) x 2(N); block tile 128 x 128.
// Wave tile: 32(M) x 64(N) = 8 accumulators -> 8 WMMA per 12 b128 loads/K-step.
// ---------------------------------------------------------------------------
__device__ __forceinline__ void store_tile(float* __restrict__ C, int N,
                                           const float* __restrict__ bias,
                                           int m0, int n0, v8f acc, int r, int hi) {
  int col = n0 + r;
  float bv = bias ? bias[col] : 0.0f;
#pragma unroll
  for (int i = 0; i < 8; ++i) {
    int row = m0 + i + 8 * hi;
    C[(size_t)row * N + col] = acc[i] + bv;
  }
}

__global__ __launch_bounds__(256) void gemm_bf16_wmma(
    const bf16* __restrict__ A, const bf16* __restrict__ W,
    const float* __restrict__ bias, float* __restrict__ C, int M, int N, int K) {
  const int lane = threadIdx.x & 31;
  const int wave = threadIdx.x >> 5;
  const int r = lane & 15;
  const int hi = lane >> 4;
  const int m0 = blockIdx.x * 128 + (wave >> 1) * 32;
  const int n0 = blockIdx.y * 128 + (wave & 1) * 64;

  v8f acc[2][4] = {{{}, {}, {}, {}}, {{}, {}, {}, {}}};

  for (int k0 = 0; k0 < K; k0 += 32) {
    // Prefetch the streaming A/W rows a few K-steps ahead (global_prefetch_b8).
    if (k0 + 256 < K) {
      __builtin_prefetch(A + (size_t)(m0 + r) * K + k0 + 256, 0, 1);
      __builtin_prefetch(W + (size_t)(n0 + r) * K + k0 + 256, 0, 1);
    }
    v16bf a0 = load_frag(A, K, m0 + r, k0, hi);
    v16bf a1 = load_frag(A, K, m0 + 16 + r, k0, hi);
    v16bf b0 = load_frag(W, K, n0 + 0 + r, k0, hi);
    v16bf b1 = load_frag(W, K, n0 + 16 + r, k0, hi);
    v16bf b2 = load_frag(W, K, n0 + 32 + r, k0, hi);
    v16bf b3 = load_frag(W, K, n0 + 48 + r, k0, hi);
    acc[0][0] = WMMA_BF16(a0, b0, acc[0][0]);
    acc[0][1] = WMMA_BF16(a0, b1, acc[0][1]);
    acc[0][2] = WMMA_BF16(a0, b2, acc[0][2]);
    acc[0][3] = WMMA_BF16(a0, b3, acc[0][3]);
    acc[1][0] = WMMA_BF16(a1, b0, acc[1][0]);
    acc[1][1] = WMMA_BF16(a1, b1, acc[1][1]);
    acc[1][2] = WMMA_BF16(a1, b2, acc[1][2]);
    acc[1][3] = WMMA_BF16(a1, b3, acc[1][3]);
  }
#pragma unroll
  for (int mi = 0; mi < 2; ++mi)
#pragma unroll
    for (int ni = 0; ni < 4; ++ni)
      store_tile(C, N, bias, m0 + mi * 16, n0 + ni * 16, acc[mi][ni], r, hi);
}

// ---------------------------------------------------------------------------
// RoPE + head repack.
// In : Qf,Kf,Vf fp32 [B,N,D] (D = H*64), cos/sin fp32 [B,N,64]
// Out: Qh,Kh bf16 [B,H,N,64];  Vt bf16 [B,H,64,N] (transposed for WMMA B-frags)
// One thread per (b,n,h,d<32); handles the rotate-half pair (d, d+32).
// ---------------------------------------------------------------------------
__global__ __launch_bounds__(256) void rope_pack_kernel(
    const float* __restrict__ Qf, const float* __restrict__ Kf,
    const float* __restrict__ Vf, const float* __restrict__ cosp,
    const float* __restrict__ sinp, bf16* __restrict__ Qh,
    bf16* __restrict__ Kh, bf16* __restrict__ Vt) {
  int idx = blockIdx.x * 256 + threadIdx.x;  // B*N*H*32 = 2M exactly
  int d = idx & 31;
  int h = (idx >> 5) & 15;
  int n = (idx >> 9) & 1023;
  int b = idx >> 19;

  size_t src = ((size_t)(b * 1024 + n)) * 1024 + h * 64 + d;
  size_t cs = ((size_t)(b * 1024 + n)) * 64 + d;
  float c0 = cosp[cs], s0 = sinp[cs];
  float c1 = cosp[cs + 32], s1 = sinp[cs + 32];

  float q0 = Qf[src], q1 = Qf[src + 32];
  float k0 = Kf[src], k1 = Kf[src + 32];
  float v0 = Vf[src], v1 = Vf[src + 32];

  float qo0 = q0 * c0 - q1 * s0;   // d < 32:  q*c - rot*s
  float qo1 = q1 * c1 + q0 * s1;   // d >= 32: q*c + rot*s
  float ko0 = k0 * c0 - k1 * s0;
  float ko1 = k1 * c1 + k0 * s1;

  size_t dst = ((size_t)((b * 16 + h) * 1024 + n)) * 64 + d;
  Qh[dst] = (bf16)qo0;
  Qh[dst + 32] = (bf16)qo1;
  Kh[dst] = (bf16)ko0;
  Kh[dst + 32] = (bf16)ko1;

  size_t vdst = ((size_t)((b * 16 + h) * 64 + d)) * 1024 + n;
  Vt[vdst] = (bf16)v0;
  Vt[vdst + (size_t)32 * 1024] = (bf16)v1;
}

// ---------------------------------------------------------------------------
// Attention: one wave (32 thr) per (b, h, 16-query tile).
// Flash-style over 4 chunks of 256 keys. Scores & P staged in ~25 KB LDS so
// ~12 single-wave workgroups fit per 320 KB WGP for latency hiding.
// Output written bf16 [B,N,D] for the final O-projection GEMM.
// ---------------------------------------------------------------------------
#define ATT_CH 256

__global__ __launch_bounds__(32) void attn_kernel(
    const bf16* __restrict__ Qh, const bf16* __restrict__ Kh,
    const bf16* __restrict__ Vt, bf16* __restrict__ Ob) {
  __shared__ float sc[16][ATT_CH + 8];  // score chunk (padded)
  __shared__ bf16 pb[16][ATT_CH];       // exp(P) bf16
  __shared__ float rmaxs[16];
  __shared__ float rsums[16];
  __shared__ float rscales[16];

  const int lane = threadIdx.x;
  const int r = lane & 15;
  const int hi = lane >> 4;

  const int qt = blockIdx.x & 63;
  const int h = (blockIdx.x >> 6) & 15;
  const int b = blockIdx.x >> 10;

  const bf16* Qp = Qh + (((size_t)(b * 16 + h)) * 1024 + qt * 16) * 64;
  const bf16* Kp = Kh + ((size_t)(b * 16 + h)) * 1024 * 64;
  const bf16* Vp = Vt + ((size_t)(b * 16 + h)) * 64 * 1024;

  if (lane < 16) {
    rmaxs[lane] = -1e30f;
    rsums[lane] = 0.0f;
  }
  __syncthreads();

  // Q-tile A fragments (rows = r, K = head dim 0..63)
  v16bf aq0 = load_frag(Qp, 64, r, 0, hi);
  v16bf aq1 = load_frag(Qp, 64, r, 32, hi);

  v8f o0 = {}, o1 = {}, o2 = {}, o3 = {};
  const float scale = 0.125f;  // 1/sqrt(64)

  for (int c = 0; c < 1024 / ATT_CH; ++c) {
    // ---- scores S = Q * K^T for keys [c*ATT_CH, +ATT_CH) ----
    for (int kt = 0; kt < ATT_CH / 16; ++kt) {
      int kb = c * ATT_CH + kt * 16;
      v16bf bk0 = load_frag(Kp, 64, kb + r, 0, hi);
      v16bf bk1 = load_frag(Kp, 64, kb + r, 32, hi);
      v8f s = {};
      s = WMMA_BF16(aq0, bk0, s);
      s = WMMA_BF16(aq1, bk1, s);
#pragma unroll
      for (int i = 0; i < 8; ++i) sc[i + 8 * hi][kt * 16 + r] = s[i] * scale;
    }
    __syncthreads();

    // ---- online softmax: lane pair (r, r+16) handles row r ----
    {
      int row = r;
      int col0 = hi * (ATT_CH / 2);
      float mx = -1e30f;
      for (int j = 0; j < ATT_CH / 2; ++j) mx = fmaxf(mx, sc[row][col0 + j]);
      mx = fmaxf(mx, __shfl_xor(mx, 16, 32));
      float oldm = rmaxs[row];
      float newm = fmaxf(oldm, mx);
      float sum = 0.0f;
      for (int j = 0; j < ATT_CH / 2; ++j) {
        float p = __expf(sc[row][col0 + j] - newm);
        pb[row][col0 + j] = (bf16)p;
        sum += p;
      }
      sum += __shfl_xor(sum, 16, 32);
      if (hi == 0) {
        float f = __expf(oldm - newm);
        rscales[row] = f;
        rsums[row] = rsums[row] * f + sum;
        rmaxs[row] = newm;
      }
    }
    __syncthreads();

    // ---- rescale accumulators, then O += P * V over this chunk ----
    float fs[8];
#pragma unroll
    for (int i = 0; i < 8; ++i) fs[i] = rscales[i + 8 * hi];
#pragma unroll
    for (int i = 0; i < 8; ++i) {
      o0[i] *= fs[i]; o1[i] *= fs[i]; o2[i] *= fs[i]; o3[i] *= fs[i];
    }

    for (int kc = 0; kc < ATT_CH / 32; ++kc) {
      v16bf ap = load_frag(&pb[0][0], ATT_CH, r, kc * 32, hi);
      int kglob = c * ATT_CH + kc * 32;
      v16bf bv0 = load_frag(Vp, 1024, 0 + r, kglob, hi);
      v16bf bv1 = load_frag(Vp, 1024, 16 + r, kglob, hi);
      v16bf bv2 = load_frag(Vp, 1024, 32 + r, kglob, hi);
      v16bf bv3 = load_frag(Vp, 1024, 48 + r, kglob, hi);
      o0 = WMMA_BF16(ap, bv0, o0);
      o1 = WMMA_BF16(ap, bv1, o1);
      o2 = WMMA_BF16(ap, bv2, o2);
      o3 = WMMA_BF16(ap, bv3, o3);
    }
    __syncthreads();
  }

  // ---- epilogue: divide by row sums, write bf16 [B, N, H*64] ----
  float rs[8];
#pragma unroll
  for (int i = 0; i < 8; ++i) rs[i] = rsums[i + 8 * hi];
#pragma unroll
  for (int i = 0; i < 8; ++i) {
    int qrow = qt * 16 + i + 8 * hi;
    size_t base = ((size_t)(b * 1024 + qrow)) * 1024 + h * 64;
    Ob[base + 0 + r] = (bf16)(o0[i] / rs[i]);
    Ob[base + 16 + r] = (bf16)(o1[i] / rs[i]);
    Ob[base + 32 + r] = (bf16)(o2[i] / rs[i]);
    Ob[base + 48 + r] = (bf16)(o3[i] / rs[i]);
  }
}

// ---------------------------------------------------------------------------
// Launch
// ---------------------------------------------------------------------------
extern "C" void kernel_launch(void* const* d_in, const int* in_sizes, int n_in,
                              void* d_out, int out_size, void* d_ws, size_t ws_size,
                              hipStream_t stream) {
  const float* hidden = (const float*)d_in[0];
  const float* cosp = (const float*)d_in[1];
  const float* sinp = (const float*)d_in[2];
  const float* wq = (const float*)d_in[3];
  const float* bq = (const float*)d_in[4];
  const float* wk = (const float*)d_in[5];
  const float* wv = (const float*)d_in[6];
  const float* bv = (const float*)d_in[7];
  const float* wo = (const float*)d_in[8];
  const float* bo = (const float*)d_in[9];

  const int M = 4096, N = 1024, K = 1024;  // M = B*Nseq

  char* ws = (char*)d_ws;
  const size_t MB = (size_t)1 << 20;
  bf16* Xb = (bf16*)(ws + 0 * MB);    // 8 MB
  bf16* Wqb = (bf16*)(ws + 8 * MB);   // 2 MB
  bf16* Wkb = (bf16*)(ws + 10 * MB);
  bf16* Wvb = (bf16*)(ws + 12 * MB);
  bf16* Wob = (bf16*)(ws + 14 * MB);
  float* Qf = (float*)(ws + 16 * MB);  // 16 MB each
  float* Kf = (float*)(ws + 32 * MB);
  float* Vf = (float*)(ws + 48 * MB);
  bf16* Qhb = (bf16*)(ws + 64 * MB);   // 8 MB each
  bf16* Khb = (bf16*)(ws + 72 * MB);
  bf16* Vtb = (bf16*)(ws + 80 * MB);
  bf16* Obb = (bf16*)(ws + 88 * MB);

  // 1) fp32 -> bf16 conversions
  cvt_f32_bf16<<<(M * K + 255) / 256, 256, 0, stream>>>(hidden, Xb, M * K);
  cvt_f32_bf16<<<(N * K + 255) / 256, 256, 0, stream>>>(wq, Wqb, N * K);
  cvt_f32_bf16<<<(N * K + 255) / 256, 256, 0, stream>>>(wk, Wkb, N * K);
  cvt_f32_bf16<<<(N * K + 255) / 256, 256, 0, stream>>>(wv, Wvb, N * K);
  cvt_f32_bf16<<<(N * K + 255) / 256, 256, 0, stream>>>(wo, Wob, N * K);

  // 2) QKV projections (WMMA GEMMs), block tile 128x128
  dim3 ggrid(M / 128, N / 128);
  gemm_bf16_wmma<<<ggrid, 256, 0, stream>>>(Xb, Wqb, bq, Qf, M, N, K);
  gemm_bf16_wmma<<<ggrid, 256, 0, stream>>>(Xb, Wkb, nullptr, Kf, M, N, K);
  gemm_bf16_wmma<<<ggrid, 256, 0, stream>>>(Xb, Wvb, bv, Vf, M, N, K);

  // 3) RoPE + repack to (B,H,N,64) / V transposed (B,H,64,N)
  rope_pack_kernel<<<(4 * 1024 * 16 * 32) / 256, 256, 0, stream>>>(
      Qf, Kf, Vf, cosp, sinp, Qhb, Khb, Vtb);

  // 4) attention (one wave per (b,h,qtile))
  attn_kernel<<<4 * 16 * 64, 32, 0, stream>>>(Qhb, Khb, Vtb, Obb);

  // 5) output projection -> fp32 d_out
  gemm_bf16_wmma<<<ggrid, 256, 0, stream>>>(Obb, Wob, bo, (float*)d_out, M, N, K);
}
